// SimpleGRU_21045339750671
// MI455X (gfx1250) — matrix operand order repaired and checked by reading, not compiled, not run on hardware
//
#include <hip/hip_runtime.h>
#include <hip/hip_bf16.h>

// ---------------------------------------------------------------------------
// SimpleGRU on MI455X (gfx1250, wave32):
//   phase 0: prep      — fp32 -> bf16 convert of x, transpose+convert weights,
//                        zero h ping-pong buffers and grid-barrier counter
//   phase 1: gates_x   — WMMA GEMM (65536x512)@(512x1536) -> fp32 gates in ws
//   phase 2: gru_scan  — persistent kernel, 1024 sequential steps:
//                        per-step GEMM h@w_h via v_wmma_f32_16x16x32_bf16,
//                        w_h^T slice in LDS, h bf16 ping-pong in global (L2),
//                        fast fp32 gate math, device-wide barrier per step
//   phase 3: y         — WMMA GEMM (65536x512)@(512x512) + bias -> d_out
// ---------------------------------------------------------------------------

typedef unsigned short u16;
typedef __attribute__((ext_vector_type(16))) __bf16        v16bf;
typedef __attribute__((ext_vector_type(8)))  float         v8f;
typedef __attribute__((ext_vector_type(4)))  unsigned int  u32x4;

union ABFrag {
    v16bf v;
    u32x4 q[2];
};

__device__ __forceinline__ u16 f2bf(float f) {
    unsigned u = __float_as_uint(f);
    unsigned r = u + 0x7FFFu + ((u >> 16) & 1u);   // round-to-nearest-even
    return (u16)(r >> 16);
}

// hardware-rate sigmoid/tanh: v_exp_f32 + v_rcp_f32, no branches, no divides
__device__ __forceinline__ float fast_sigmoid(float x) {
    return __builtin_amdgcn_rcpf(1.0f + __expf(-x));
}
__device__ __forceinline__ float fast_tanh(float x) {
    return 1.0f - 2.0f * __builtin_amdgcn_rcpf(1.0f + __expf(2.0f * x));
}

// --------------------------- phase 0: prep ---------------------------------
__global__ void k_prep(const float* __restrict__ x,
                       const float* __restrict__ wi,
                       const float* __restrict__ wh,
                       const float* __restrict__ wo,
                       u16* __restrict__ xbf,
                       u16* __restrict__ wiT,
                       u16* __restrict__ whT,
                       u16* __restrict__ woT,
                       u16* __restrict__ hbf,
                       unsigned* __restrict__ cnt) {
    const size_t nth = (size_t)gridDim.x * blockDim.x;
    const size_t id  = (size_t)blockIdx.x * blockDim.x + threadIdx.x;

    for (size_t i = id; i < (size_t)64 * 1024 * 512; i += nth)
        xbf[i] = f2bf(x[i]);

    // w_i, w_h: (512,1536) -> transposed bf16 (1536,512)
    for (size_t i = id; i < (size_t)1536 * 512; i += nth) {
        const size_t g = i >> 9, k = i & 511;
        wiT[i] = f2bf(wi[k * 1536 + g]);
        whT[i] = f2bf(wh[k * 1536 + g]);
    }
    // w_out: (512,512) -> transposed bf16
    for (size_t i = id; i < (size_t)512 * 512; i += nth) {
        const size_t o = i >> 9, k = i & 511;
        woT[i] = f2bf(wo[k * 512 + o]);
    }
    // zero both h ping-pong buffers (h0 = 0)
    for (size_t i = id; i < (size_t)2 * 64 * 512; i += nth)
        hbf[i] = 0;
    if (id == 0) *cnt = 0u;
}

// ----------------------- phase 1: gates_x GEMM -----------------------------
// gates[t][b][g] = sum_k x[b][t][k] * w_i[k][g];  rows r = t*64 + b
__global__ __launch_bounds__(256) void k_gemm_gates(const u16* __restrict__ xbf,
                                                    const u16* __restrict__ wiT,
                                                    float* __restrict__ gates) {
    const int tid  = threadIdx.x;
    const int wave = tid >> 5;
    const int lane = tid & 31;
    const int n16  = lane & 15;
    const int half = lane >> 4;
    const int mtile = blockIdx.x * 8 + wave;            // 4096 M-tiles

    const int rA = mtile * 16 + n16;                    // A-row for this lane
    const int tA = rA >> 6, bA = rA & 63;
    const u16* arow = xbf + ((size_t)bA * 1024 + tA) * 512;

    for (int nc = 0; nc < 24; ++nc) {                   // 96 N-tiles, 4 at a time
        const int g0 = nc * 64;
        v8f c[4] = {};
        for (int kt = 0; kt < 16; ++kt) {
            const int k0 = kt * 32;
            ABFrag a;
            a.q[0] = *(const u32x4*)(arow + k0 + half * 16);
            a.q[1] = *(const u32x4*)(arow + k0 + half * 16 + 8);
#pragma unroll
            for (int j = 0; j < 4; ++j) {
                ABFrag bf;
                const u16* bp = wiT + (size_t)(g0 + j * 16 + n16) * 512 + k0 + half * 16;
                bf.q[0] = ((const u32x4*)bp)[0];
                bf.q[1] = ((const u32x4*)bp)[1];
                c[j] = __builtin_amdgcn_wmma_f32_16x16x32_bf16(
                    false, a.v, false, bf.v, (short)0, c[j], false, false);
            }
        }
#pragma unroll
        for (int j = 0; j < 4; ++j) {
            const int gcol = g0 + j * 16 + n16;
#pragma unroll
            for (int v = 0; v < 8; ++v) {
                const int r2 = mtile * 16 + half * 8 + v;
                const int t2 = r2 >> 6, b2 = r2 & 63;
                gates[((size_t)t2 * 64 + b2) * 1536 + gcol] = c[j][v];
            }
        }
    }
}

// ------------------------- phase 2: GRU scan -------------------------------
#define NWG2 32
__global__ __launch_bounds__(128) void k_gru_scan(const float* __restrict__ gates,
                                                  const float* __restrict__ bias,
                                                  const u16* __restrict__ whT,
                                                  u16* __restrict__ hbf,
                                                  u16* __restrict__ outs,
                                                  unsigned* __restrict__ cnt,
                                                  float* __restrict__ hfinal) {
    __shared__ u16 wlds[48 * 512];                      // 48 KB: z/r/a col-slices of w_h^T

    const int tid  = threadIdx.x;
    const int wg   = blockIdx.x;                        // owns H-columns [16*wg, 16*wg+16)
    const int wave = tid >> 5;
    const int lane = tid & 31;
    const int n16  = lane & 15;
    const int half = lane >> 4;
    const int col0 = wg * 16;

    // preload w_h^T slices into LDS: LDS row j -> global row (j/16)*512 + col0 + (j%16)
    {
        u32x4* dst = (u32x4*)wlds;
        const u32x4* src = (const u32x4*)whT;
        for (int idx = tid; idx < 48 * 64; idx += 128) {
            const int row  = idx >> 6;
            const int c    = idx & 63;
            const int gate = row >> 4;
            const int nn   = row & 15;
            dst[idx] = src[(size_t)(gate * 512 + col0 + nn) * 64 + c];
        }
    }
    __syncthreads();

    const int gz = col0 + n16;                          // this lane's h column
    const float bz = bias[gz];
    const float br = bias[512 + gz];
    const float ba = bias[1024 + gz];
    const int mrow0 = wave * 16;                        // this wave's batch rows
    v8f hprev = {};                                     // fp32 h slice, h0 = 0

    for (int t = 0; t < 1024; ++t) {
        const u16* hread  = hbf + (size_t)(t & 1) * (64 * 512);
        u16*       hwrite = hbf + (size_t)((t + 1) & 1) * (64 * 512);
        const u16* arow   = hread + (size_t)(mrow0 + n16) * 512;

        v8f cz = {}, cr = {}, ca = {};
        for (int kt = 0; kt < 16; ++kt) {
            const int k0 = kt * 32;
            ABFrag a;
            a.q[0] = *(const u32x4*)(arow + k0 + half * 16);
            a.q[1] = *(const u32x4*)(arow + k0 + half * 16 + 8);

            ABFrag fz, fr, fa;
            const u16* pz = wlds + (size_t)(0  + n16) * 512 + k0 + half * 16;
            const u16* pr = wlds + (size_t)(16 + n16) * 512 + k0 + half * 16;
            const u16* pa = wlds + (size_t)(32 + n16) * 512 + k0 + half * 16;
            fz.q[0] = ((const u32x4*)pz)[0]; fz.q[1] = ((const u32x4*)pz)[1];
            fr.q[0] = ((const u32x4*)pr)[0]; fr.q[1] = ((const u32x4*)pr)[1];
            fa.q[0] = ((const u32x4*)pa)[0]; fa.q[1] = ((const u32x4*)pa)[1];

            cz = __builtin_amdgcn_wmma_f32_16x16x32_bf16(false, a.v, false, fz.v, (short)0, cz, false, false);
            cr = __builtin_amdgcn_wmma_f32_16x16x32_bf16(false, a.v, false, fr.v, (short)0, cr, false, false);
            ca = __builtin_amdgcn_wmma_f32_16x16x32_bf16(false, a.v, false, fa.v, (short)0, ca, false, false);
        }

        // elementwise GRU update on this lane's (8 rows x 1 col) of the tile
#pragma unroll
        for (int v = 0; v < 8; ++v) {
            const int m = mrow0 + half * 8 + v;
            const size_t gbase = ((size_t)t * 64 + m) * 1536 + gz;
            const float zx = gates[gbase];
            const float rx = gates[gbase + 512];
            const float ax = gates[gbase + 1024];
            const float z  = fast_sigmoid(zx + cz[v] + bz);
            const float r  = fast_sigmoid(rx + cr[v] + br);
            const float aa = fast_tanh(ax + r * ca[v] + ba);
            const float hn = (1.0f - z) * hprev[v] + z * aa;
            hprev[v] = hn;
            const u16 hb = f2bf(hn);
            hwrite[(size_t)m * 512 + gz] = hb;
            outs[((size_t)t * 64 + m) * 512 + gz] = hb;
        }

        // device-wide barrier: all 32 WGs must publish h before next step
        __threadfence();
        __syncthreads();
        if (tid == 0) {
            atomicAdd(cnt, 1u);
            const unsigned tgt = (unsigned)(t + 1) * NWG2;
            while (atomicAdd(cnt, 0u) < tgt) { __builtin_amdgcn_s_sleep(2); }
        }
        __syncthreads();
        __threadfence();
    }

    // h_final (fp32) -> tail of d_out
#pragma unroll
    for (int v = 0; v < 8; ++v) {
        const int m = mrow0 + half * 8 + v;
        hfinal[(size_t)m * 512 + gz] = hprev[v];
    }
}

// --------------------------- phase 3: output GEMM --------------------------
// y[b][t][o] = sum_h outs[t][b][h] * w_out[h][o] + b_out[o]; rows r' = b*1024+t
__global__ __launch_bounds__(256) void k_gemm_out(const u16* __restrict__ outsbf,
                                                  const u16* __restrict__ woT,
                                                  const float* __restrict__ bout,
                                                  float* __restrict__ y) {
    const int tid  = threadIdx.x;
    const int wave = tid >> 5;
    const int lane = tid & 31;
    const int n16  = lane & 15;
    const int half = lane >> 4;
    const int mtile = blockIdx.x * 8 + wave;            // 4096 M-tiles

    const int rp = mtile * 16 + n16;
    const int bI = rp >> 10, tI = rp & 1023;
    const u16* arow = outsbf + ((size_t)tI * 64 + bI) * 512;

    for (int nc = 0; nc < 8; ++nc) {                    // 32 N-tiles, 4 at a time
        const int o0 = nc * 64;
        v8f c[4] = {};
        for (int kt = 0; kt < 16; ++kt) {
            const int k0 = kt * 32;
            ABFrag a;
            a.q[0] = *(const u32x4*)(arow + k0 + half * 16);
            a.q[1] = *(const u32x4*)(arow + k0 + half * 16 + 8);
#pragma unroll
            for (int j = 0; j < 4; ++j) {
                ABFrag bf;
                const u16* bp = woT + (size_t)(o0 + j * 16 + n16) * 512 + k0 + half * 16;
                bf.q[0] = ((const u32x4*)bp)[0];
                bf.q[1] = ((const u32x4*)bp)[1];
                c[j] = __builtin_amdgcn_wmma_f32_16x16x32_bf16(
                    false, a.v, false, bf.v, (short)0, c[j], false, false);
            }
        }
#pragma unroll
        for (int j = 0; j < 4; ++j) {
            const int col = o0 + j * 16 + n16;
            const float bb = bout[col];
#pragma unroll
            for (int v = 0; v < 8; ++v) {
                const int rr = mtile * 16 + half * 8 + v;
                y[(size_t)rr * 512 + col] = c[j][v] + bb;
            }
        }
    }
}

// ------------------------------- launcher ----------------------------------
extern "C" void kernel_launch(void* const* d_in, const int* in_sizes, int n_in,
                              void* d_out, int out_size, void* d_ws, size_t ws_size,
                              hipStream_t stream) {
    (void)in_sizes; (void)n_in; (void)out_size; (void)ws_size;

    const float* x     = (const float*)d_in[0];   // (64,1024,512)
    const float* w_i   = (const float*)d_in[1];   // (512,1536)
    const float* w_h   = (const float*)d_in[2];   // (512,1536)
    const float* b     = (const float*)d_in[3];   // (1536,)
    const float* w_out = (const float*)d_in[4];   // (512,512)
    const float* b_out = (const float*)d_in[5];   // (512,)

    float* y      = (float*)d_out;                           // (64,1024,512)
    float* hfinal = y + (size_t)64 * 1024 * 512;             // (64,512)

    // workspace carve-up (~517 MB)
    char* ws = (char*)d_ws;
    size_t off = 0;
    auto take = [&](size_t bytes) -> char* {
        char* p = ws + off;
        off = (off + bytes + 255) & ~(size_t)255;
        return p;
    };
    unsigned* cnt  = (unsigned*)take(256);
    u16* xbf  = (u16*)take((size_t)64 * 1024 * 512 * 2);     // 64 MB
    u16* wiT  = (u16*)take((size_t)1536 * 512 * 2);          // 1.5 MB
    u16* whT  = (u16*)take((size_t)1536 * 512 * 2);          // 1.5 MB
    u16* woT  = (u16*)take((size_t)512 * 512 * 2);           // 0.5 MB
    u16* hbf  = (u16*)take((size_t)2 * 64 * 512 * 2);        // 128 KB ping-pong
    u16* outsb = (u16*)take((size_t)1024 * 64 * 512 * 2);    // 64 MB
    float* gates = (float*)take((size_t)1024 * 64 * 1536 * 4); // 384 MB

    k_prep<<<1024, 256, 0, stream>>>(x, w_i, w_h, w_out, xbf, wiT, whT, woT, hbf, cnt);
    k_gemm_gates<<<512, 256, 0, stream>>>(xbf, wiT, gates);
    k_gru_scan<<<NWG2, 128, 0, stream>>>(gates, b, whT, hbf, outsb, cnt, hfinal);
    k_gemm_out<<<512, 256, 0, stream>>>(outsb, woT, b_out, y);
}
